// MCActor_6743098655484
// MI455X (gfx1250) — matrix operand order repaired and checked by reading, not compiled, not run on hardware
//
#include <hip/hip_runtime.h>
#include <hip/hip_bf16.h>
#include <math.h>

typedef __attribute__((ext_vector_type(16))) __bf16        v16bf;
typedef __attribute__((ext_vector_type(8)))  float         v8f;
typedef __attribute__((ext_vector_type(4)))  unsigned int  v4u;

#define HH   256          // H
#define NN   1024         // N
#define RT   32           // rows per block tile (2 WMMA row-tiles)
#define HPAD 264          // bf16 LDS row stride (pad to dodge bank conflicts)
#define SPAD 264          // f32  LDS row stride

union U16 { v16bf v; unsigned short u[16]; };

__device__ __forceinline__ unsigned short f2bf(float f) {
    __bf16 h = (__bf16)f;
    return __builtin_bit_cast(unsigned short, h);
}

// A fragment (16x32 bf16) from an LDS-staged bf16 tile, ISA 7.12.2 layout.
// Each 8-element half is k-contiguous -> compiler merges into ds_load_b128.
__device__ __forceinline__ v16bf load_afrag(const unsigned short* HID, int hs,
                                            int ks, int lane) {
    U16 f;
    const int m  = lane & 15;
    const int kb = (lane >> 4) ? 8 : 0;
#pragma unroll
    for (int j = 0; j < 8; ++j) {
        const int k0 = ks + kb + ((j < 4) ? 2 * j : 2 * j + 8);
        f.u[2 * j]     = HID[m * hs + k0];
        f.u[2 * j + 1] = HID[m * hs + k0 + 1];
    }
    return f.v;
}

// B fragment (32x16 bf16) pre-packed in register order:
// Wpk[((step*16 + ct)*32 + lane)*16 + e], 32 contiguous bytes per lane
// -> two global_load_b128, fully coalesced across the wave.
__device__ __forceinline__ v16bf load_bfrag_pk(const unsigned short* __restrict__ Wpk,
                                               int step, int ct, int lane) {
    const v4u* p = (const v4u*)(Wpk + ((((step << 4) + ct) << 5) + lane) * 16);
    U16 f;
    *(v4u*)(&f.u[0]) = p[0];
    *(v4u*)(&f.u[8]) = p[1];
    return f.v;
}

#define WMMA_BF16(A, B, C) \
    __builtin_amdgcn_wmma_f32_16x16x32_bf16(false, (A), false, (B), (short)0, (C), false, false)

// Recompute one 32-row sn_h tile (rows n0..n0+31 of batch b) into LDS (fp32).
// hid1 = relu(sn_input @ sn_w1 + b1) staged bf16 -> GEMM1 (WMMA, 2x2 register
// blocking: two row-tiles share each B fragment) -> + b2.
__device__ __forceinline__ void compute_snh_tile32(
    int b, int n0,
    const float* __restrict__ sn_input, const float* __restrict__ sn_w1,
    const float* __restrict__ sn_b1, const unsigned short* __restrict__ w2pk,
    const float* __restrict__ sn_b2,
    unsigned short* hidL /*RT*HPAD*/, float* snhL /*RT*SPAD*/) {
    const int t = threadIdx.x;
    // hidden layer: K=6 is trivial; thread t owns column t across RT rows
#pragma unroll 4
    for (int r = 0; r < RT; ++r) {
        const long row = (long)b * NN + n0 + r;
        float s = sn_b1[t];
#pragma unroll
        for (int i = 0; i < 6; ++i)
            s += sn_input[row * 6 + i] * sn_w1[i * HH + t];
        hidL[r * HPAD + t] = f2bf(fmaxf(s, 0.f));
    }
    __syncthreads();

    const int wave = t >> 5, lane = t & 31;
    v8f acc00 = {}, acc01 = {}, acc10 = {}, acc11 = {};
#pragma unroll
    for (int st = 0; st < 8; ++st) {
        if (st < 7)
            __builtin_prefetch(w2pk + (((((st + 1) << 4) + wave * 2) << 5) + lane) * 16, 0, 3);
        v16bf a0 = load_afrag(hidL,             HPAD, st * 32, lane);
        v16bf a1 = load_afrag(hidL + 16 * HPAD, HPAD, st * 32, lane);
        v16bf b0 = load_bfrag_pk(w2pk, st, wave * 2,     lane);
        v16bf b1 = load_bfrag_pk(w2pk, st, wave * 2 + 1, lane);
        acc00 = WMMA_BF16(a0, b0, acc00);
        acc01 = WMMA_BF16(a0, b1, acc01);
        acc10 = WMMA_BF16(a1, b0, acc10);
        acc11 = WMMA_BF16(a1, b1, acc11);
    }
    const int mb = (lane >> 4) ? 8 : 0;
    const int c0 = wave * 32 + (lane & 15), c1 = c0 + 16;
    const float bb0 = sn_b2[c0], bb1 = sn_b2[c1];
#pragma unroll
    for (int r = 0; r < 8; ++r) {
        snhL[(mb + r) * SPAD + c0]        = acc00[r] + bb0;
        snhL[(mb + r) * SPAD + c1]        = acc01[r] + bb1;
        snhL[(mb + r + 16) * SPAD + c0]   = acc10[r] + bb0;
        snhL[(mb + r + 16) * SPAD + c1]   = acc11[r] + bb1;
    }
    __syncthreads();
}

// ---------------- kernels ----------------

// Weight prep: convert to bf16 AND permute into WMMA B-fragment register order.
//   out[i], i = (((step*16 + ct)*32 + lane)*16 + e)
//   k = step*32 + (lane>=16 ? 8 : 0) + e + (e>=8 ? 8 : 0) ; n = ct*16 + (lane&15)
// w2pk   from sn_w2[k*H + n]           (GEMM1: sn_h = hid1 @ sn_w2)
// wsnTpk from att_W[n*2H + k]          (GEMM2: spre = sn_h @ att_W[:, :H]^T)
__global__ void k_prep(const float* __restrict__ sn_w2, const float* __restrict__ att_W,
                       unsigned short* __restrict__ w2pk, unsigned short* __restrict__ wsnTpk) {
    const int i = blockIdx.x * 256 + threadIdx.x;  // 65536 total
    const int e = i & 15, lane = (i >> 4) & 31, ct = (i >> 9) & 15, step = i >> 13;
    const int n = ct * 16 + (lane & 15);
    const int k = step * 32 + ((lane >> 4) ? 8 : 0) + e + ((e >= 8) ? 8 : 0);
    w2pk[i]   = f2bf(sn_w2[k * HH + n]);
    wsnTpk[i] = f2bf(att_W[n * (2 * HH) + k]);
}

// mc encoder: mc_h = relu(mc_input@mc_w1 + b1) @ mc_w2 + b2, one block per batch
__global__ void k_mc(const float* __restrict__ mc_input, const float* __restrict__ mc_w1,
                     const float* __restrict__ mc_b1, const float* __restrict__ mc_w2,
                     const float* __restrict__ mc_b2, float* __restrict__ mc_h) {
    __shared__ float hid[HH];
    const int b = blockIdx.x, t = threadIdx.x;
    float s = mc_b1[t];
#pragma unroll
    for (int i = 0; i < 4; ++i) s += mc_input[b * 4 + i] * mc_w1[i * HH + t];
    hid[t] = fmaxf(s, 0.f);
    __syncthreads();
    float o = mc_b2[t];
    for (int k = 0; k < HH; ++k) o += hid[k] * mc_w2[k * HH + t];
    mc_h[b * HH + t] = o;
}

// mc_pre[b,h] = sum_k mc_h[b,k] * att_W[h, H+k]  (batch-constant attention term)
__global__ void k_mcpre(const float* __restrict__ mc_h, const float* __restrict__ att_W,
                        float* __restrict__ mc_pre) {
    __shared__ float m[HH];
    const int b = blockIdx.x, t = threadIdx.x;
    m[t] = mc_h[b * HH + t];
    __syncthreads();
    float s = 0.f;
    for (int k = 0; k < HH; ++k) s += m[k] * att_W[t * (2 * HH) + HH + k];
    mc_pre[b * HH + t] = s;
}

__global__ void k_zero(float* __restrict__ p, int n) {
    const int i = blockIdx.x * 256 + threadIdx.x;
    if (i < n) p[i] = 0.f;
}

// fused sn-encode + attention logits for a 32-row tile
__global__ void k_sn_att(const float* __restrict__ sn_input, const float* __restrict__ sn_w1,
                         const float* __restrict__ sn_b1, const unsigned short* __restrict__ w2pk,
                         const float* __restrict__ sn_b2, const unsigned short* __restrict__ wsnTpk,
                         const float* __restrict__ mc_pre, const float* __restrict__ att_v,
                         float* __restrict__ logits) {
    __shared__ unsigned short hidL[RT * HPAD];
    __shared__ float          snhL[RT * SPAD];
    __shared__ unsigned short snbL[RT * HPAD];
    __shared__ float          red[RT];
    const int b = blockIdx.y, n0 = blockIdx.x * RT, t = threadIdx.x;

    compute_snh_tile32(b, n0, sn_input, sn_w1, sn_b1, w2pk, sn_b2, hidL, snhL);

    if (t < RT) red[t] = 0.f;
#pragma unroll 4
    for (int r = 0; r < RT; ++r) snbL[r * HPAD + t] = f2bf(snhL[r * SPAD + t]);
    __syncthreads();

    const int wave = t >> 5, lane = t & 31;
    v8f acc00 = {}, acc01 = {}, acc10 = {}, acc11 = {};
#pragma unroll
    for (int st = 0; st < 8; ++st) {
        if (st < 7)
            __builtin_prefetch(wsnTpk + (((((st + 1) << 4) + wave * 2) << 5) + lane) * 16, 0, 3);
        v16bf a0 = load_afrag(snbL,             HPAD, st * 32, lane);
        v16bf a1 = load_afrag(snbL + 16 * HPAD, HPAD, st * 32, lane);
        v16bf b0 = load_bfrag_pk(wsnTpk, st, wave * 2,     lane);
        v16bf b1 = load_bfrag_pk(wsnTpk, st, wave * 2 + 1, lane);
        acc00 = WMMA_BF16(a0, b0, acc00);
        acc01 = WMMA_BF16(a0, b1, acc01);
        acc10 = WMMA_BF16(a1, b0, acc10);
        acc11 = WMMA_BF16(a1, b1, acc11);
    }
    const int mb = (lane >> 4) ? 8 : 0;
    const int c0 = wave * 32 + (lane & 15), c1 = c0 + 16;
    const float av0 = att_v[c0], av1 = att_v[c1];
    const float mp0 = mc_pre[b * HH + c0], mp1 = mc_pre[b * HH + c1];
#pragma unroll
    for (int r = 0; r < 8; ++r) {
        const float p0 = tanhf(acc00[r] + mp0) * av0 + tanhf(acc01[r] + mp1) * av1;
        const float p1 = tanhf(acc10[r] + mp0) * av0 + tanhf(acc11[r] + mp1) * av1;
        atomicAdd(&red[mb + r], p0);
        atomicAdd(&red[mb + r + 16], p1);
    }
    __syncthreads();
    if (t < RT) logits[(long)b * NN + n0 + t] = red[t];
}

// softmax over N=1024 per batch (256 threads x 4 values)
__global__ void k_softmax(const float* __restrict__ logits, float* __restrict__ attn) {
    __shared__ float s[256];
    const int b = blockIdx.x, t = threadIdx.x;
    float v[4], mx = -INFINITY;
#pragma unroll
    for (int j = 0; j < 4; ++j) {
        v[j] = logits[(long)b * NN + t + j * 256];
        mx = fmaxf(mx, v[j]);
    }
    s[t] = mx; __syncthreads();
    for (int o = 128; o; o >>= 1) { if (t < o) s[t] = fmaxf(s[t], s[t + o]); __syncthreads(); }
    mx = s[0]; __syncthreads();
    float e[4], sum = 0.f;
#pragma unroll
    for (int j = 0; j < 4; ++j) { e[j] = __expf(v[j] - mx); sum += e[j]; }
    s[t] = sum; __syncthreads();
    for (int o = 128; o; o >>= 1) { if (t < o) s[t] += s[t + o]; __syncthreads(); }
    const float inv = 1.f / s[0];
#pragma unroll
    for (int j = 0; j < 4; ++j) attn[(long)b * NN + t + j * 256] = e[j] * inv;
}

// context[b,h] += sum_r attn[b,n0+r] * sn_h[b,n0+r,h]  (tile recomputed in LDS)
__global__ void k_context(const float* __restrict__ sn_input, const float* __restrict__ sn_w1,
                          const float* __restrict__ sn_b1, const unsigned short* __restrict__ w2pk,
                          const float* __restrict__ sn_b2, const float* __restrict__ attn,
                          float* __restrict__ context) {
    __shared__ unsigned short hidL[RT * HPAD];
    __shared__ float          snhL[RT * SPAD];
    __shared__ float          aw[RT];
    const int b = blockIdx.y, n0 = blockIdx.x * RT, t = threadIdx.x;
    if (t < RT) aw[t] = attn[(long)b * NN + n0 + t];   // covered by tile's first barrier
    compute_snh_tile32(b, n0, sn_input, sn_w1, sn_b1, w2pk, sn_b2, hidL, snhL);
    float acc = 0.f;
#pragma unroll
    for (int r = 0; r < RT; ++r) acc += aw[r] * snhL[r * SPAD + t];
    atomicAdd(&context[b * HH + t], acc);
}

// pointer MLP: out = relu(relu([mc_h, context]@fc1 + b1)@fc2 + b2)
__global__ void k_pointer(const float* __restrict__ mc_h, const float* __restrict__ context,
                          const float* __restrict__ fc1_w, const float* __restrict__ fc1_b,
                          const float* __restrict__ fc2_w, const float* __restrict__ fc2_b,
                          float* __restrict__ outv) {
    __shared__ float inp[2 * HH];
    __shared__ float hid[HH];
    const int b = blockIdx.x, t = threadIdx.x;
    inp[t]      = mc_h[b * HH + t];
    inp[HH + t] = context[b * HH + t];
    __syncthreads();
    float s = fc1_b[t];
    for (int k = 0; k < 2 * HH; ++k) s += inp[k] * fc1_w[k * HH + t];
    hid[t] = fmaxf(s, 0.f);
    __syncthreads();
    float o = fc2_b[t];
    for (int k = 0; k < HH; ++k) o += hid[k] * fc2_w[k * HH + t];
    outv[b * HH + t] = fmaxf(o, 0.f);
}

// probs[b,n] = sum_h tanh(sn_h[b,n,h] + out[b,h]) * ptr_v[h]
__global__ void k_probs(const float* __restrict__ sn_input, const float* __restrict__ sn_w1,
                        const float* __restrict__ sn_b1, const unsigned short* __restrict__ w2pk,
                        const float* __restrict__ sn_b2, const float* __restrict__ outv,
                        const float* __restrict__ ptr_v, float* __restrict__ probs) {
    __shared__ unsigned short hidL[RT * HPAD];
    __shared__ float          snhL[RT * SPAD];
    __shared__ float          ov[HH];
    __shared__ float          pv[HH];
    __shared__ float          rowsum[RT];
    const int b = blockIdx.y, n0 = blockIdx.x * RT, t = threadIdx.x;
    ov[t] = outv[b * HH + t];
    pv[t] = ptr_v[t];
    if (t < RT) rowsum[t] = 0.f;                      // covered by tile's first barrier
    compute_snh_tile32(b, n0, sn_input, sn_w1, sn_b1, w2pk, sn_b2, hidL, snhL);
    const int r = t >> 3, h0 = t & 7;                 // 8 threads per row
    float p = 0.f;
#pragma unroll
    for (int j = 0; j < 32; ++j) {
        const int h = h0 + j * 8;
        p += tanhf(snhL[r * SPAD + h] + ov[h]) * pv[h];
    }
    atomicAdd(&rowsum[r], p);
    __syncthreads();
    if (t < RT) probs[(long)b * NN + n0 + t] = rowsum[t];
}

extern "C" void kernel_launch(void* const* d_in, const int* in_sizes, int n_in,
                              void* d_out, int out_size, void* d_ws, size_t ws_size,
                              hipStream_t stream) {
    const float* mc_input = (const float*)d_in[0];
    const float* sn_input = (const float*)d_in[1];
    const float* mc_w1    = (const float*)d_in[2];
    const float* mc_b1    = (const float*)d_in[3];
    const float* mc_w2    = (const float*)d_in[4];
    const float* mc_b2    = (const float*)d_in[5];
    const float* sn_w1    = (const float*)d_in[6];
    const float* sn_b1    = (const float*)d_in[7];
    const float* sn_w2    = (const float*)d_in[8];
    const float* sn_b2    = (const float*)d_in[9];
    const float* att_v    = (const float*)d_in[10];
    const float* att_W    = (const float*)d_in[11];
    const float* ptr_v    = (const float*)d_in[12];
    const float* fc1_w    = (const float*)d_in[13];
    const float* fc1_b    = (const float*)d_in[14];
    const float* fc2_w    = (const float*)d_in[15];
    const float* fc2_b    = (const float*)d_in[16];
    (void)in_sizes; (void)n_in; (void)out_size; (void)ws_size;

    char* ws = (char*)d_ws;
    float*          logits  = (float*)(ws);                              // 1 MB
    float*          attn    = (float*)(ws + (1u << 20));                 // 1 MB
    float*          mc_h    = (float*)(ws + (2u << 20));                 // 256 KB
    float*          mc_pre  = (float*)(ws + (2u << 20) + (256u << 10));  // 256 KB
    float*          context = (float*)(ws + (2u << 20) + (512u << 10));  // 256 KB
    float*          outv    = (float*)(ws + (2u << 20) + (768u << 10));  // 256 KB
    unsigned short* w2pk    = (unsigned short*)(ws + (3u << 20));        // 128 KB
    unsigned short* wsnTpk  = (unsigned short*)(ws + (3u << 20) + (128u << 10)); // 128 KB

    const dim3 blk(256);
    const dim3 big(NN / RT, 256);  // (n-tiles, batches)

    k_prep   <<<256, blk, 0, stream>>>(sn_w2, att_W, w2pk, wsnTpk);
    k_mc     <<<256, blk, 0, stream>>>(mc_input, mc_w1, mc_b1, mc_w2, mc_b2, mc_h);
    k_mcpre  <<<256, blk, 0, stream>>>(mc_h, att_W, mc_pre);
    k_zero   <<<(256 * HH + 255) / 256, blk, 0, stream>>>(context, 256 * HH);
    k_sn_att <<<big, blk, 0, stream>>>(sn_input, sn_w1, sn_b1, w2pk, sn_b2,
                                       wsnTpk, mc_pre, att_v, logits);
    k_softmax<<<256, blk, 0, stream>>>(logits, attn);
    k_context<<<big, blk, 0, stream>>>(sn_input, sn_w1, sn_b1, w2pk, sn_b2, attn, context);
    k_pointer<<<256, blk, 0, stream>>>(mc_h, context, fc1_w, fc1_b, fc2_w, fc2_b, outv);
    k_probs  <<<big, blk, 0, stream>>>(sn_input, sn_w1, sn_b1, w2pk, sn_b2,
                                       outv, ptr_v, (float*)d_out);
}
